// MultiHeadAttention_65068754534553
// MI455X (gfx1250) — matrix operand order repaired and checked by reading, not compiled, not run on hardware
//
#include <hip/hip_runtime.h>
#include <hip/hip_bf16.h>

// ---------------------------------------------------------------------------
// CDNA5 (gfx1250) banded multi-head attention, bf16 WMMA everywhere.
//   B=2, S=2048, IN=1024, E=1024, H=16, HD=64, WINDOW=256 (pad=128)
// Stage plan:
//   1) f32->bf16 casts of x / Wqkv / Wo
//   2) WMMA GEMM x@Wqkv^T (+bias, padding mask) -> q,k,v [B,H,S,64] bf16
//      (K-slab stage-in via Tensor Data Mover when the builtin exists,
//       double-buffered in LDS; cooperative b128 loads otherwise)
//   3) banded attention (17 key tiles, f32 softmax in LDS, WMMA for QK^T & PV)
//   4) WMMA GEMM vals@Wo^T + bo -> f32 out
// ---------------------------------------------------------------------------

typedef __bf16 bf16_t;
typedef __attribute__((ext_vector_type(16))) __bf16 v16bf;
typedef __attribute__((ext_vector_type(8)))  float  v8f;

#define WMMA_BF16(a, b, c) \
  __builtin_amdgcn_wmma_f32_16x16x32_bf16(false, (a), false, (b), (short)0, (c), false, false)

#if __has_builtin(__builtin_amdgcn_tensor_load_to_lds) && \
    __has_builtin(__builtin_amdgcn_s_wait_tensorcnt)
#define USE_TDM 1
#endif

#ifdef USE_TDM
typedef __attribute__((ext_vector_type(4))) unsigned int u32x4;
typedef __attribute__((ext_vector_type(8))) int          i32x8;
typedef __attribute__((ext_vector_type(4))) int          i32x4;

// 2D TDM load: tile (tile0 x tile1) of 2-byte elements from row-major tensor
// (row stride = stride0 elements) into contiguous LDS at lds_off.
// D# packing per CDNA5 ISA 8.3/8.4 (group0: count/lds/global/type,
// group1: data_size, tensor dims, tile dims, dim0 stride).
__device__ __forceinline__ void tdm_load_2d(unsigned lds_off, const bf16_t* gsrc,
                                            unsigned dim0, unsigned dim1,
                                            unsigned stride0,
                                            unsigned tile0, unsigned tile1) {
  unsigned long long ga = (unsigned long long)(uintptr_t)gsrc;
  u32x4 g0;
  g0[0] = 1u;                                            // count=1, user desc
  g0[1] = lds_off;                                       // LDS byte address
  g0[2] = (unsigned)(ga & 0xFFFFFFFFu);                  // global_addr[31:0]
  g0[3] = (unsigned)((ga >> 32) & 0x01FFFFFFu) | (2u << 30);  // [56:32] | type=2
  i32x8 g1;
  g1[0] = 0x00010000;                                    // data_size=1 (2B), mask=0
  g1[1] = (int)((dim0 & 0xFFFFu) << 16);                 // tensor_dim0[15:0]
  g1[2] = (int)(((dim0 >> 16) & 0xFFFFu) | ((dim1 & 0xFFFFu) << 16));
  g1[3] = (int)(((dim1 >> 16) & 0xFFFFu) | ((tile0 & 0xFFFFu) << 16));
  g1[4] = (int)(tile1 & 0xFFFFu);                        // tile_dim1 (tile_dim2=0)
  g1[5] = (int)stride0;                                  // tensor_dim0_stride[31:0]
  g1[6] = 0;                                             // stride0 hi / stride1 lo
  g1[7] = 0;
  i32x4 z4 = {0, 0, 0, 0};
#if defined(__clang_major__) && (__clang_major__ >= 23)
  i32x8 z8 = {0, 0, 0, 0, 0, 0, 0, 0};
  __builtin_amdgcn_tensor_load_to_lds(g0, g1, z4, z4, z8, 0);
#else
  __builtin_amdgcn_tensor_load_to_lds(g0, g1, z4, z4, 0);
#endif
}
#endif  // USE_TDM

static constexpr int   Bsz  = 2;
static constexpr int   Ssz  = 2048;
static constexpr int   INd  = 1024;
static constexpr int   EMB  = 1024;
static constexpr int   HH   = 16;
static constexpr int   HD   = 64;
static constexpr float NEGF = -9.0e15f;

// --------------------------------------------------------------------------
// f32 -> bf16 conversion
// --------------------------------------------------------------------------
__global__ __launch_bounds__(256) void cvt_f32_bf16(const float* __restrict__ src,
                                                    bf16_t* __restrict__ dst, int n) {
  int i = blockIdx.x * 256 + threadIdx.x;
  if (i < n) dst[i] = (bf16_t)src[i];
}

// --------------------------------------------------------------------------
// Generic WMMA GEMM: C[M,N] = A[M,K](bf16) * W[N,K]^T(bf16) + bias
//   mode 0: QKV epilogue (padding mask, scatter to q/k/v [B,H,S,64] bf16)
//   mode 1: output projection epilogue (f32 store)
// Block = 256 threads (8 waves), tile 64(M) x 64(N), K-step 32.
// --------------------------------------------------------------------------
__global__ __launch_bounds__(256)
void gemm_wmma(const bf16_t* __restrict__ A, const bf16_t* __restrict__ W,
               const float* __restrict__ bias, int K, int mode,
               const int* __restrict__ pmask,
               bf16_t* __restrict__ qp, bf16_t* __restrict__ kp, bf16_t* __restrict__ vp,
               float* __restrict__ outp)
{
#ifdef USE_TDM
  __shared__ __align__(16) bf16_t As[2][64 * 32];   // double-buffered for TDM
  __shared__ __align__(16) bf16_t Bs[2][64 * 32];
#else
  __shared__ __align__(16) bf16_t As[1][64 * 32];
  __shared__ __align__(16) bf16_t Bs[1][64 * 32];
#endif

  const int t    = threadIdx.x;
  const int lane = t & 31;
  const int w    = t >> 5;
  const int wr   = w >> 1;       // 0..3 (M sub-tile)
  const int wc   = w & 1;        // 0..1 (N sub-tile, 32 cols each)
  const int mrow = blockIdx.y * 64;
  const int ncol = blockIdx.x * 64;

  const int ln15 = lane & 15;
  const int kb8  = (lane >> 4) * 8;
  const int kb16 = (lane >> 4) * 16;

  v8f c0 = {}, c1 = {};

#ifdef USE_TDM
  // ---- TDM double-buffered pipeline: wave 0 drives the Tensor Data Mover ----
  const unsigned aoff0 = (unsigned)(uintptr_t)&As[0][0];
  const unsigned aoff1 = (unsigned)(uintptr_t)&As[1][0];
  const unsigned boff0 = (unsigned)(uintptr_t)&Bs[0][0];
  const unsigned boff1 = (unsigned)(uintptr_t)&Bs[1][0];
  if (w == 0) {
    tdm_load_2d(aoff0, A + (size_t)mrow * K, (unsigned)K, 64u, (unsigned)K, 32u, 64u);
    tdm_load_2d(boff0, W + (size_t)ncol * K, (unsigned)K, 64u, (unsigned)K, 32u, 64u);
    __builtin_amdgcn_s_wait_tensorcnt((short)0);
  }
  __syncthreads();
  int cur = 0;
  for (int k0 = 0; k0 < K; k0 += 32) {
    const bool more = (k0 + 32 < K);
    if (w == 0 && more) {   // prefetch next slab into the other buffer
      tdm_load_2d(cur ? aoff0 : aoff1, A + (size_t)mrow * K + k0 + 32,
                  (unsigned)K, 64u, (unsigned)K, 32u, 64u);
      tdm_load_2d(cur ? boff0 : boff1, W + (size_t)ncol * K + k0 + 32,
                  (unsigned)K, 64u, (unsigned)K, 32u, 64u);
    }
    v16bf a, b0, b1;
#pragma unroll
    for (int i = 0; i < 8; ++i) {
      a[i]     = As[cur][(wr * 16 + ln15) * 32 + kb16 + i];
      a[8 + i] = As[cur][(wr * 16 + ln15) * 32 + kb16 + 8 + i];
    }
#pragma unroll
    for (int i = 0; i < 16; ++i) {
      b0[i] = Bs[cur][(wc * 32 + ln15) * 32 + kb16 + i];
      b1[i] = Bs[cur][(wc * 32 + 16 + ln15) * 32 + kb16 + i];
    }
    c0 = WMMA_BF16(a, b0, c0);
    c1 = WMMA_BF16(a, b1, c1);
    if (w == 0 && more) __builtin_amdgcn_s_wait_tensorcnt((short)0);
    __syncthreads();
    cur ^= 1;
  }
#else
  // ---- Fallback: cooperative b128 loads + prefetch ----
  const int lrow = t >> 2;        // 0..63
  const int lcol = (t & 3) * 8;   // 0,8,16,24
  for (int k0 = 0; k0 < K; k0 += 32) {
    __syncthreads();
    *(uint4*)&As[0][lrow * 32 + lcol] =
        *(const uint4*)&A[(size_t)(mrow + lrow) * K + k0 + lcol];
    *(uint4*)&Bs[0][lrow * 32 + lcol] =
        *(const uint4*)&W[(size_t)(ncol + lrow) * K + k0 + lcol];
    if (k0 + 32 < K) {
      __builtin_prefetch(&A[(size_t)(mrow + lrow) * K + k0 + 32 + lcol], 0, 1);
      __builtin_prefetch(&W[(size_t)(ncol + lrow) * K + k0 + 32 + lcol], 0, 1);
    }
    __syncthreads();
    v16bf a, b0, b1;
#pragma unroll
    for (int i = 0; i < 8; ++i) {
      a[i]     = As[0][(wr * 16 + ln15) * 32 + kb16 + i];
      a[8 + i] = As[0][(wr * 16 + ln15) * 32 + kb16 + 8 + i];
    }
#pragma unroll
    for (int i = 0; i < 16; ++i) {
      b0[i] = Bs[0][(wc * 32 + ln15) * 32 + kb16 + i];
      b1[i] = Bs[0][(wc * 32 + 16 + ln15) * 32 + kb16 + i];
    }
    c0 = WMMA_BF16(a, b0, c0);
    c1 = WMMA_BF16(a, b1, c1);
  }
#endif

  const int mB = (lane >> 4) * 8;
  float cacc[2][8];
#pragma unroll
  for (int r = 0; r < 8; ++r) { cacc[0][r] = c0[r]; cacc[1][r] = c1[r]; }

#pragma unroll
  for (int tt = 0; tt < 2; ++tt) {
#pragma unroll
    for (int r = 0; r < 8; ++r) {
      int gm = mrow + wr * 16 + mB + r;
      int gn = ncol + wc * 32 + tt * 16 + ln15;
      float v = cacc[tt][r] + bias[gn];
      if (mode == 0) {
        int bidx = gm >> 11;         // / S
        int s    = gm & 2047;        // % S
        if (pmask[bidx * Ssz + s] != 0) v = 0.0f;
        int h = gn / 192;
        int c = gn - h * 192;
        bf16_t* dst = (c < 64) ? qp : (c < 128 ? kp : vp);
        int d = (c < 64) ? c : (c < 128 ? c - 64 : c - 128);
        dst[((size_t)(bidx * HH + h) * Ssz + s) * HD + d] = (bf16_t)v;
      } else {
        outp[(size_t)gm * EMB + gn] = v;
      }
    }
  }
}

// --------------------------------------------------------------------------
// Banded attention: block = 64 threads (2 waves), each wave = 16 queries.
// Band per 16-query tile = 272 keys (17 x 16), padded to 288 for K=32 WMMA.
// --------------------------------------------------------------------------
__global__ __launch_bounds__(64)
void banded_attn(const bf16_t* __restrict__ Q, const bf16_t* __restrict__ Kmat,
                 const bf16_t* __restrict__ Vmat, bf16_t* __restrict__ vals)
{
  __shared__ float sc[2][16][288];
  __shared__ float rowsum[2][16];

  const int w     = threadIdx.x >> 5;
  const int lane  = threadIdx.x & 31;
  const int nq32  = Ssz / 32;                  // 64 query-pair tiles
  const int qt    = blockIdx.x % nq32;
  const int bh    = blockIdx.x / nq32;         // b*H + h
  const int b     = bh / HH;
  const int h     = bh - b * HH;
  const int qbase = qt * 32 + w * 16;

  const bf16_t* Qp = Q    + (size_t)bh * Ssz * HD;
  const bf16_t* Kp = Kmat + (size_t)bh * Ssz * HD;
  const bf16_t* Vp = Vmat + (size_t)bh * Ssz * HD;

  const int ln15 = lane & 15;
  const int kb8  = (lane >> 4) * 8;
  const int kb16 = (lane >> 4) * 16;
  const int mB   = (lane >> 4) * 8;

  // ---- Q fragments (A-matrix, d = 0..31 and 32..63) ----
  v16bf aq0, aq1;
  {
    const bf16_t* qr = Qp + (size_t)(qbase + ln15) * HD;
#pragma unroll
    for (int i = 0; i < 8; ++i) {
      aq0[i]     = qr[kb16 + i];
      aq0[8 + i] = qr[kb16 + 8 + i];
      aq1[i]     = qr[32 + kb16 + i];
      aq1[8 + i] = qr[32 + kb16 + 8 + i];
    }
  }

  // ---- scores: 17 key tiles of 16 ----
  for (int kt = 0; kt < 17; ++kt) {
    int kb  = qbase - 128 + kt * 16;
    int key = kb + ln15;
    bool okr = (key >= 0) && (key < Ssz);
    const bf16_t* kr = Kp + (size_t)(okr ? key : 0) * HD;
    v16bf bk0, bk1;
#pragma unroll
    for (int i = 0; i < 16; ++i) {
      bk0[i] = okr ? kr[kb16 + i]      : (bf16_t)0.0f;
      bk1[i] = okr ? kr[32 + kb16 + i] : (bf16_t)0.0f;
    }
    v8f sa = {};
    sa = WMMA_BF16(aq0, bk0, sa);
    sa = WMMA_BF16(aq1, bk1, sa);
#pragma unroll
    for (int r = 0; r < 8; ++r) {
      int iq = qbase + mB + r;
      int j  = kb + ln15;
      bool band = (j >= 0) && (j < Ssz) && (iq - j <= 128) && (j - iq <= 128);
      sc[w][mB + r][kt * 16 + ln15] = (band ? sa[r] : NEGF) * 0.125f;  // scale AFTER fill
    }
  }
  __syncthreads();

  // ---- softmax (2 lanes per row, 272 cols; exp in place, fold 1/sum later) ----
  {
    int r     = ln15;
    int half  = lane >> 4;
    int cbase = half * 136;
    float mx = -3.0e38f;
    for (int c = 0; c < 136; ++c) mx = fmaxf(mx, sc[w][r][cbase + c]);
    mx = fmaxf(mx, __shfl_xor(mx, 16, 32));
    float sum = 0.0f;
    for (int c = 0; c < 136; ++c) {
      float p = __expf(sc[w][r][cbase + c] - mx);
      sum += p;
      sc[w][r][cbase + c] = p;
    }
    sum += __shfl_xor(sum, 16, 32);
    rowsum[w][r] = sum;
    for (int c = 0; c < 8; ++c) sc[w][r][272 + half * 8 + c] = 0.0f;  // zero pad cols
  }
  __syncthreads();

  // ---- P @ V : 9 K-steps of 32, 4 N-tiles of 16 (d = 0..63) ----
  v8f acc[4] = {};
  for (int ks = 0; ks < 9; ++ks) {
    v16bf ap;
#pragma unroll
    for (int i = 0; i < 8; ++i) {
      ap[i]     = (bf16_t)sc[w][ln15][ks * 32 + kb16 + i];
      ap[8 + i] = (bf16_t)sc[w][ln15][ks * 32 + kb16 + 8 + i];
    }
    int krow0 = qbase - 128 + ks * 32 + kb16;
#pragma unroll
    for (int nt = 0; nt < 4; ++nt) {
      v16bf bv;
      int dcol = nt * 16 + ln15;
#pragma unroll
      for (int i = 0; i < 16; ++i) {
        int kk = krow0 + i;
        bv[i] = (kk >= 0 && kk < Ssz) ? Vp[(size_t)kk * HD + dcol] : (bf16_t)0.0f;
      }
      acc[nt] = WMMA_BF16(ap, bv, acc[nt]);
    }
  }

  // ---- epilogue: divide by rowsum, store vals[B,S,E] bf16 ----
#pragma unroll
  for (int nt = 0; nt < 4; ++nt) {
#pragma unroll
    for (int r = 0; r < 8; ++r) {
      int row = qbase + mB + r;
      int d   = nt * 16 + ln15;
      float o = acc[nt][r] / rowsum[w][mB + r];
      vals[((size_t)(b * Ssz + row)) * EMB + h * HD + d] = (bf16_t)o;
    }
  }
}

// --------------------------------------------------------------------------
// Launch
// --------------------------------------------------------------------------
extern "C" void kernel_launch(void* const* d_in, const int* in_sizes, int n_in,
                              void* d_out, int out_size, void* d_ws, size_t ws_size,
                              hipStream_t stream)
{
  (void)in_sizes; (void)n_in; (void)out_size; (void)ws_size;
  const float* x    = (const float*)d_in[0];
  const int*   pm   = (const int*)d_in[1];
  const float* Wqkv = (const float*)d_in[2];
  const float* bqkv = (const float*)d_in[3];
  const float* Wo   = (const float*)d_in[4];
  const float* bo   = (const float*)d_in[5];
  float* out = (float*)d_out;

  const size_t MS = (size_t)Bsz * Ssz;          // 4096 rows
  char* ws = (char*)d_ws;
  size_t off = 0;
  bf16_t* xbf    = (bf16_t*)(ws + off); off += MS * INd * 2;              // 8 MB
  bf16_t* wqkvbf = (bf16_t*)(ws + off); off += (size_t)3 * EMB * INd * 2; // 6 MB
  bf16_t* wobf   = (bf16_t*)(ws + off); off += (size_t)EMB * EMB * 2;     // 2 MB
  bf16_t* qbuf   = (bf16_t*)(ws + off); off += MS * EMB * 2;              // 8 MB
  bf16_t* kbuf   = (bf16_t*)(ws + off); off += MS * EMB * 2;
  bf16_t* vbuf   = (bf16_t*)(ws + off); off += MS * EMB * 2;
  bf16_t* valsb  = (bf16_t*)(ws + off); off += MS * EMB * 2;

  int nx = (int)(MS * INd);
  int nw = 3 * EMB * INd;
  int no = EMB * EMB;
  cvt_f32_bf16<<<(nx + 255) / 256, 256, 0, stream>>>(x, xbf, nx);
  cvt_f32_bf16<<<(nw + 255) / 256, 256, 0, stream>>>(Wqkv, wqkvbf, nw);
  cvt_f32_bf16<<<(no + 255) / 256, 256, 0, stream>>>(Wo, wobf, no);

  dim3 g1(3 * EMB / 64, (unsigned)(MS / 64));   // 48 x 64
  gemm_wmma<<<g1, 256, 0, stream>>>(xbf, wqkvbf, bqkv, INd, 0,
                                    pm, qbuf, kbuf, vbuf, nullptr);

  banded_attn<<<Bsz * HH * (Ssz / 32), 64, 0, stream>>>(qbuf, kbuf, vbuf, valsb);

  dim3 g2(EMB / 64, (unsigned)(MS / 64));       // 16 x 64
  gemm_wmma<<<g2, 256, 0, stream>>>(valsb, wobf, bo, EMB, 1,
                                    nullptr, nullptr, nullptr, nullptr, out);
}
